// SnnNetwork_59777354826394
// MI455X (gfx1250) — matrix operand order repaired and checked by reading, not compiled, not run on hardware
//
#include <hip/hip_runtime.h>

// Problem sizes (match reference)
#define Bsz    4096
#define INsz   1024
#define H0sz   4000
#define Rsz    2000
#define H1sz   4000
#define OUTsz  10
#define RP     2016   // Rsz padded to multiple of 32 (K dim)
#define H0P    4096   // H0 padded to multiple of 64 (N dim, layer-1 weights)
#define RNP    2048   // R padded to multiple of 64 (N dim, recurrent weights)

// Block tiling
#define BM     128
#define BN     64
#define KS     32
#define LPAD   40                 // LDS row stride in bf16 elems (20 banks -> conflict-free)
#define ASZ    (BM * LPAD)        // 5120 elems per A buffer
#define BSZ    (BN * LPAD)        // 2560 elems per B buffer

typedef __attribute__((ext_vector_type(16))) __bf16 v16bf;
typedef __attribute__((ext_vector_type(8)))  float  v8f;
typedef __attribute__((ext_vector_type(4)))  int    v4i_;

__device__ __forceinline__ v16bf as_v16bf(uint4 lo, uint4 hi) {
    union { uint4 u[2]; v16bf v; } x;
    x.u[0] = lo; x.u[1] = hi;
    return x.v;
}

__device__ __forceinline__ float sigmoidf_(float x) {
    return 1.0f / (1.0f + __expf(-x));
}

__device__ __forceinline__ unsigned short f2bf(float x) {
    unsigned int u = __float_as_uint(x);
    u += 0x7FFFu + ((u >> 16) & 1u);   // round-to-nearest-even
    return (unsigned short)(u >> 16);
}

// ---------------------------------------------------------------------------
// CDNA5 async global->LDS copy (ASYNCcnt path). Guarded: falls back to
// register staging if the toolchain lacks the builtin.
// Builtin prototype (from hipcc diagnostic): arg0 = v4i addrspace(1)*,
// arg1 = v4i addrspace(3)*, then imm offset + imm cpol.
// ---------------------------------------------------------------------------
#if __has_builtin(__builtin_amdgcn_global_load_async_to_lds_b128)
#define HAVE_ASYNC_LDS 1
typedef __attribute__((address_space(1))) v4i_* gv4p;
typedef __attribute__((address_space(3))) v4i_* lv4p;
__device__ __forceinline__ void cp_async_b128(const unsigned short* g, unsigned short* l) {
    __builtin_amdgcn_global_load_async_to_lds_b128((gv4p)(g), (lv4p)(l), 0, 0);
}
__device__ __forceinline__ void wait_async0() {
#if __has_builtin(__builtin_amdgcn_s_wait_asynccnt)
    __builtin_amdgcn_s_wait_asynccnt(0);
#else
    asm volatile("s_wait_asynccnt 0x0" ::: "memory");
#endif
}
#else
#define HAVE_ASYNC_LDS 0
#endif

// ---------------------------------------------------------------------------
// fp32 -> bf16 with zero padding in both rows (N) and cols (K).
// ---------------------------------------------------------------------------
__global__ void k_cvt_pad(const float* __restrict__ src, int srcStride, int srcOff,
                          unsigned short* __restrict__ dst, int rowsCopy,
                          int Kcopy, int Kdst, int total) {
    int t = blockIdx.x * blockDim.x + threadIdx.x;
    if (t >= total) return;
    int r = t / Kdst;
    int k = t - r * Kdst;
    unsigned short v = 0;
    if (r < rowsCopy && k < Kcopy) v = f2bf(src[(size_t)r * srcStride + srcOff + k]);
    dst[t] = v;
}

// ---------------------------------------------------------------------------
// LDS-tiled, double-buffered accumulation of a 128x64 block tile of
// C += A(MxK,row-major bf16) * B(NxK,row-major bf16)^T.
// Each of the 8 waves (4M x 2N) owns a 32x32 register tile:
//   c[0]=(m0,n0) c[1]=(m0,n1) c[2]=(m1,n0) c[3]=(m1,n1)
// Staging uses GLOBAL_LOAD_ASYNC_TO_LDS_B128 when available.
// ---------------------------------------------------------------------------
__device__ __forceinline__ void gemm_tile_acc(v8f c[4],
                                              const unsigned short* __restrict__ A, int lda,
                                              const unsigned short* __restrict__ B, int ldb,
                                              int K, int blockM, int blockN,
                                              unsigned short* As, unsigned short* Bs) {
    const int tid  = threadIdx.x;
    const int lane = tid & 31;
    const int half = lane >> 4;
    const int l16  = lane & 15;
    const int wv   = tid >> 5;
    const int wvM  = wv & 3;
    const int wvN  = wv >> 2;

    // global->LDS: A tile = 128 rows x 32 elems = 512 16B-chunks (2/thread);
    //              B tile =  64 rows x 32 elems = 256 16B-chunks (1/thread)
    const int arow = tid >> 2;
    const int acol = (tid & 3) * 8;
    const unsigned short* gA0 = A + (size_t)(blockM + arow)      * lda + acol;
    const unsigned short* gA1 = A + (size_t)(blockM + arow + 64) * lda + acol;
    const unsigned short* gB  = B + (size_t)(blockN + arow)      * ldb + acol;
    unsigned short* sA0 = As + arow * LPAD + acol;
    unsigned short* sA1 = As + (arow + 64) * LPAD + acol;
    unsigned short* sB  = Bs + arow * LPAD + acol;

    // fragment read pointers (A: chunks at +0 / +16 ; B: chunks at +0 / +16)
    const unsigned short* fA0 = As + (wvM * 32 + l16)      * LPAD + half * 8;
    const unsigned short* fA1 = As + (wvM * 32 + 16 + l16) * LPAD + half * 8;
    const unsigned short* fB0 = Bs + (wvN * 32 + l16)      * LPAD + half * 8;
    const unsigned short* fB1 = Bs + (wvN * 32 + 16 + l16) * LPAD + half * 8;

    const int steps = K >> 5;

    // prologue: stage K-step 0 into buffer 0
#if HAVE_ASYNC_LDS
    cp_async_b128(gA0, sA0);
    cp_async_b128(gA1, sA1);
    cp_async_b128(gB,  sB);
    wait_async0();
    __syncthreads();
#else
    {
        uint4 ra0 = *(const uint4*)gA0;
        uint4 ra1 = *(const uint4*)gA1;
        uint4 rb  = *(const uint4*)gB;
        *(uint4*)sA0 = ra0;
        *(uint4*)sA1 = ra1;
        *(uint4*)sB  = rb;
    }
    __syncthreads();
#endif

    for (int ks = 0; ks < steps; ++ks) {
        const int cur  = ks & 1;
        const int nxt  = cur ^ 1;
        const bool more = (ks + 1) < steps;
#if HAVE_ASYNC_LDS
        if (more) {
            const int k0 = (ks + 1) << 5;
            cp_async_b128(gA0 + k0, sA0 + nxt * ASZ);
            cp_async_b128(gA1 + k0, sA1 + nxt * ASZ);
            cp_async_b128(gB + k0,  sB + nxt * BSZ);
        }
#else
        uint4 ra0, ra1, rb;
        if (more) {
            const int k0 = (ks + 1) << 5;
            ra0 = *(const uint4*)(gA0 + k0);
            ra1 = *(const uint4*)(gA1 + k0);
            rb  = *(const uint4*)(gB + k0);
        }
#endif
        const unsigned short* pa0 = fA0 + cur * ASZ;
        const unsigned short* pa1 = fA1 + cur * ASZ;
        const unsigned short* pb0 = fB0 + cur * BSZ;
        const unsigned short* pb1 = fB1 + cur * BSZ;
        v16bf a0 = as_v16bf(*(const uint4*)pa0, *(const uint4*)(pa0 + 16));
        v16bf a1 = as_v16bf(*(const uint4*)pa1, *(const uint4*)(pa1 + 16));
        v16bf b0 = as_v16bf(*(const uint4*)pb0, *(const uint4*)(pb0 + 16));
        v16bf b1 = as_v16bf(*(const uint4*)pb1, *(const uint4*)(pb1 + 16));
        c[0] = __builtin_amdgcn_wmma_f32_16x16x32_bf16(false, a0, false, b0, (short)0, c[0], false, false);
        c[1] = __builtin_amdgcn_wmma_f32_16x16x32_bf16(false, a0, false, b1, (short)0, c[1], false, false);
        c[2] = __builtin_amdgcn_wmma_f32_16x16x32_bf16(false, a1, false, b0, (short)0, c[2], false, false);
        c[3] = __builtin_amdgcn_wmma_f32_16x16x32_bf16(false, a1, false, b1, (short)0, c[3], false, false);
#if HAVE_ASYNC_LDS
        if (more) wait_async0();      // own async writes done; barrier makes global
        __syncthreads();
#else
        if (more) {
            *(uint4*)(sA0 + nxt * ASZ) = ra0;
            *(uint4*)(sA1 + nxt * ASZ) = ra1;
            *(uint4*)(sB + nxt * BSZ)  = rb;
        }
        __syncthreads();
#endif
    }
}

// ---------------------------------------------------------------------------
// Fused SNN membrane update for one v8f fragment.
// element i -> row (mbase + i + 8*half), column `col` (already global).
// ---------------------------------------------------------------------------
__device__ __forceinline__ void mem_update_epi(v8f c, int mbase, int col, int ld,
                                               float bias, float tm, float tad,
                                               const float* __restrict__ memO,
                                               const float* __restrict__ spkO,
                                               const float* __restrict__ bO,
                                               float* __restrict__ memN,
                                               float* __restrict__ spkN,
                                               float* __restrict__ bN,
                                               unsigned short* spkbf, int ldbf) {
    const int half = (threadIdx.x & 31) >> 4;
#pragma unroll
    for (int i = 0; i < 8; ++i) {
        int m = mbase + i + 8 * half;
        size_t idx = (size_t)m * ld + col;
        float inp = c[i] + bias;
        float mo = memO[idx], so = spkO[idx], bo = bO[idx];
        float bn = tad * bo + (1.0f - tad) * so;
        float th = 0.1f + 1.8f * bn;                               // BASE_THRE + BETA*b
        float mn = mo * tm + (1.0f - tm) * 3.0f * inp - th * so;   // R_M = 3
        float sn = (mn - th) > 0.0f ? 1.0f : 0.0f;
        memN[idx] = mn;
        spkN[idx] = sn;
        bN[idx]   = bn;
        if (spkbf)
            spkbf[(size_t)m * ldbf + col] =
                (mn - th) > 0.0f ? (unsigned short)0x3F80u : (unsigned short)0u;
    }
}

// ---------------------------------------------------------------------------
// Layer 1: inp1 = x_t @ fc_w.T + fc_b ; mem_update -> mem1n/spk1n/b1n (+bf16)
// ---------------------------------------------------------------------------
__global__ void __launch_bounds__(256)
k_gemm1(const unsigned short* __restrict__ Abf, const unsigned short* __restrict__ Wbf,
        const float* __restrict__ fcb,
        const float* __restrict__ mem1, const float* __restrict__ spk1,
        const float* __restrict__ b1,
        const float* __restrict__ tau_m1, const float* __restrict__ tau_adp1,
        float* __restrict__ mem1n, float* __restrict__ spk1n, float* __restrict__ b1n,
        unsigned short* __restrict__ spk1bf) {
    __shared__ __align__(16) unsigned short As[2 * ASZ];
    __shared__ __align__(16) unsigned short Bs[2 * BSZ];
    const int bM = blockIdx.x & 31;      // 4096/128
    const int bN = blockIdx.x >> 5;      // 4096/64
    const int blockM = bM * BM, blockN = bN * BN;
    v8f c[4] = {};
    gemm_tile_acc(c, Abf, INsz, Wbf, INsz, INsz, blockM, blockN, As, Bs);

    const int l16 = threadIdx.x & 15;
    const int wv  = threadIdx.x >> 5;
    const int mb0 = blockM + (wv & 3) * 32;
    const int nb0 = blockN + (wv >> 2) * 32;
#pragma unroll
    for (int t = 0; t < 2; ++t) {                 // two 16-col subtiles
        int col = nb0 + t * 16 + l16;
        if (col < H0sz) {
            float bias = fcb[col];
            float tm  = sigmoidf_(tau_m1[col]);
            float tad = sigmoidf_(tau_adp1[col]);
            mem_update_epi(c[t],     mb0,      col, H0sz, bias, tm, tad, mem1, spk1, b1,
                           mem1n, spk1n, b1n, spk1bf, H0sz);
            mem_update_epi(c[t + 2], mb0 + 16, col, H0sz, bias, tm, tad, mem1, spk1, b1,
                           mem1n, spk1n, b1n, spk1bf, H0sz);
        }
    }
}

// ---------------------------------------------------------------------------
// r_out_ = spk_out @ rec4out.T + spk_in @ in2out.T + biases ; cols [0,R)
// ---------------------------------------------------------------------------
__global__ void __launch_bounds__(256)
k_rout(const unsigned short* __restrict__ spkout_bf, const unsigned short* __restrict__ spkin_bf,
       const unsigned short* __restrict__ rec4out_bf, const unsigned short* __restrict__ in2out_bf,
       const float* __restrict__ rec4out_b, const float* __restrict__ in2out_b,
       const float* __restrict__ mem2, const float* __restrict__ spk2,
       const float* __restrict__ b2,
       const float* __restrict__ tau_m2, const float* __restrict__ tau_adp2,
       float* __restrict__ mem2n, float* __restrict__ spk2n, float* __restrict__ b2n) {
    __shared__ __align__(16) unsigned short As[2 * ASZ];
    __shared__ __align__(16) unsigned short Bs[2 * BSZ];
    const int bM = blockIdx.x & 31;
    const int bN = blockIdx.x >> 5;      // 2048/64
    const int blockM = bM * BM, blockN = bN * BN;
    v8f c[4] = {};
    gemm_tile_acc(c, spkout_bf, RP, rec4out_bf, RP, RP, blockM, blockN, As, Bs);
    gemm_tile_acc(c, spkin_bf,  RP, in2out_bf,  RP, RP, blockM, blockN, As, Bs);

    const int l16 = threadIdx.x & 15;
    const int wv  = threadIdx.x >> 5;
    const int mb0 = blockM + (wv & 3) * 32;
    const int nb0 = blockN + (wv >> 2) * 32;
#pragma unroll
    for (int t = 0; t < 2; ++t) {
        int n = nb0 + t * 16 + l16;
        if (n < Rsz) {
            int h = n;                                 // r_out_ -> cols [0,R)
            float bias = rec4out_b[n] + in2out_b[n];
            float tm  = sigmoidf_(tau_m2[h]);
            float tad = sigmoidf_(tau_adp2[h]);
            mem_update_epi(c[t],     mb0,      h, H1sz, bias, tm, tad, mem2, spk2, b2,
                           mem2n, spk2n, b2n, nullptr, 0);
            mem_update_epi(c[t + 2], mb0 + 16, h, H1sz, bias, tm, tad, mem2, spk2, b2,
                           mem2n, spk2n, b2n, nullptr, 0);
        }
    }
}

// ---------------------------------------------------------------------------
// r_in_ = spk1n @ x2in.T + spk_in @ rec4in.T + spk_out @ out2in.T + biases
// cols [R, 2R)
// ---------------------------------------------------------------------------
__global__ void __launch_bounds__(256)
k_rin(const unsigned short* __restrict__ spk1n_bf, const unsigned short* __restrict__ x2in_bf,
      const unsigned short* __restrict__ spkin_bf, const unsigned short* __restrict__ rec4in_bf,
      const unsigned short* __restrict__ spkout_bf, const unsigned short* __restrict__ out2in_bf,
      const float* __restrict__ x2in_b, const float* __restrict__ rec4in_b,
      const float* __restrict__ out2in_b,
      const float* __restrict__ mem2, const float* __restrict__ spk2,
      const float* __restrict__ b2,
      const float* __restrict__ tau_m2, const float* __restrict__ tau_adp2,
      float* __restrict__ mem2n, float* __restrict__ spk2n, float* __restrict__ b2n) {
    __shared__ __align__(16) unsigned short As[2 * ASZ];
    __shared__ __align__(16) unsigned short Bs[2 * BSZ];
    const int bM = blockIdx.x & 31;
    const int bN = blockIdx.x >> 5;
    const int blockM = bM * BM, blockN = bN * BN;
    v8f c[4] = {};
    gemm_tile_acc(c, spk1n_bf, H0sz, x2in_bf,   H0sz, H0sz, blockM, blockN, As, Bs);
    gemm_tile_acc(c, spkin_bf, RP,   rec4in_bf, RP,   RP,   blockM, blockN, As, Bs);
    gemm_tile_acc(c, spkout_bf, RP,  out2in_bf, RP,   RP,   blockM, blockN, As, Bs);

    const int l16 = threadIdx.x & 15;
    const int wv  = threadIdx.x >> 5;
    const int mb0 = blockM + (wv & 3) * 32;
    const int nb0 = blockN + (wv >> 2) * 32;
#pragma unroll
    for (int t = 0; t < 2; ++t) {
        int n = nb0 + t * 16 + l16;
        if (n < Rsz) {
            int h = Rsz + n;                           // r_in_ -> cols [R,2R)
            float bias = x2in_b[n] + rec4in_b[n] + out2in_b[n];
            float tm  = sigmoidf_(tau_m2[h]);
            float tad = sigmoidf_(tau_adp2[h]);
            mem_update_epi(c[t],     mb0,      h, H1sz, bias, tm, tad, mem2, spk2, b2,
                           mem2n, spk2n, b2n, nullptr, 0);
            mem_update_epi(c[t + 2], mb0 + 16, h, H1sz, bias, tm, tad, mem2, spk2, b2,
                           mem2n, spk2n, b2n, nullptr, 0);
        }
    }
}

// ---------------------------------------------------------------------------
// Output head
// ---------------------------------------------------------------------------
__global__ void k_outsum(const float* __restrict__ spk2n, const float* __restrict__ mem_out,
                         const float* __restrict__ out_tau_m, float* __restrict__ mem_out_n) {
    int t = blockIdx.x * blockDim.x + threadIdx.x;
    if (t >= Bsz * OUTsz) return;
    int row = t / OUTsz;
    int j   = t - row * OUTsz;
    const float* p = spk2n + (size_t)row * H1sz + j * 200;
    float s = 0.0f;
    for (int k = 0; k < 200; ++k) s += p[k];
    float mo = mem_out[t];
    float sg = sigmoidf_(out_tau_m[j]);
    mem_out_n[t] = mo + (s - mo) * sg;
}

__global__ void k_logsm(const float* __restrict__ mem_out_n, float* __restrict__ log_sm) {
    int row = blockIdx.x * blockDim.x + threadIdx.x;
    if (row >= Bsz) return;
    float v[OUTsz];
    float mx = -3.0e38f;
    for (int j = 0; j < OUTsz; ++j) {
        v[j] = mem_out_n[row * OUTsz + j];
        mx = fmaxf(mx, v[j]);
    }
    float s = 0.0f;
    for (int j = 0; j < OUTsz; ++j) s += __expf(v[j] - mx);
    float lse = mx + __logf(s);
    for (int j = 0; j < OUTsz; ++j) log_sm[row * OUTsz + j] = v[j] - lse;
}

// ---------------------------------------------------------------------------
extern "C" void kernel_launch(void* const* d_in, const int* in_sizes, int n_in,
                              void* d_out, int out_size, void* d_ws, size_t ws_size,
                              hipStream_t stream) {
    (void)in_sizes; (void)n_in; (void)out_size; (void)ws_size;

    const float* x_t       = (const float*)d_in[0];
    const float* mem1      = (const float*)d_in[1];
    const float* spk1      = (const float*)d_in[2];
    const float* b1        = (const float*)d_in[3];
    const float* mem2      = (const float*)d_in[4];
    const float* spk2      = (const float*)d_in[5];
    const float* b2        = (const float*)d_in[6];
    const float* mem_out   = (const float*)d_in[7];
    const float* fc_w      = (const float*)d_in[8];
    const float* fc_b      = (const float*)d_in[9];
    const float* tau_adp1  = (const float*)d_in[10];
    const float* tau_m1    = (const float*)d_in[11];
    const float* x2in_w    = (const float*)d_in[12];
    const float* x2in_b    = (const float*)d_in[13];
    const float* rec4in_w  = (const float*)d_in[14];
    const float* rec4in_b  = (const float*)d_in[15];
    const float* in2out_w  = (const float*)d_in[16];
    const float* in2out_b  = (const float*)d_in[17];
    const float* rec4out_w = (const float*)d_in[18];
    const float* rec4out_b = (const float*)d_in[19];
    const float* out2in_w  = (const float*)d_in[20];
    const float* out2in_b  = (const float*)d_in[21];
    const float* tau_adp2  = (const float*)d_in[22];
    const float* tau_m2    = (const float*)d_in[23];
    const float* out_tau_m = (const float*)d_in[24];

    // d_out layout: log_sm, mem1n, spk1n, b1n, mem2n, spk2n, b2n, mem_out_n
    float* out       = (float*)d_out;
    float* o_logsm   = out;
    float* o_mem1n   = out + (size_t)Bsz * OUTsz;
    float* o_spk1n   = o_mem1n + (size_t)Bsz * H0sz;
    float* o_b1n     = o_spk1n + (size_t)Bsz * H0sz;
    float* o_mem2n   = o_b1n   + (size_t)Bsz * H0sz;
    float* o_spk2n   = o_mem2n + (size_t)Bsz * H1sz;
    float* o_b2n     = o_spk2n + (size_t)Bsz * H1sz;
    float* o_memoutn = o_b2n   + (size_t)Bsz * H1sz;

    // Workspace: bf16 copies, padded in K (->2016) and N (->4096/2048)
    char* ws = (char*)d_ws;
    unsigned short* ws_xt      = (unsigned short*)(ws + 0);          // 4096x1024
    unsigned short* ws_fcw     = (unsigned short*)(ws + 8388608);    // 4096x1024 (4000 rows real)
    unsigned short* ws_x2in    = (unsigned short*)(ws + 16777216);   // 2048x4000
    unsigned short* ws_rec4in  = (unsigned short*)(ws + 33161216);   // 2048x2016
    unsigned short* ws_in2out  = (unsigned short*)(ws + 41418752);
    unsigned short* ws_rec4out = (unsigned short*)(ws + 49676288);
    unsigned short* ws_out2in  = (unsigned short*)(ws + 57933824);
    unsigned short* ws_spkin   = (unsigned short*)(ws + 66191360);   // 4096x2016
    unsigned short* ws_spkout  = (unsigned short*)(ws + 82706432);
    unsigned short* ws_spk1n   = (unsigned short*)(ws + 99221504);   // 4096x4000

    const int TPB = 256;
    auto nblk = [](int total, int tpb) { return (total + tpb - 1) / tpb; };

    // --- bf16 conversions (weights converted once, reused by every tile) ---
    k_cvt_pad<<<nblk(Bsz * INsz, TPB), TPB, 0, stream>>>(x_t, INsz, 0, ws_xt, Bsz, INsz, INsz, Bsz * INsz);
    k_cvt_pad<<<nblk(H0P * INsz, TPB), TPB, 0, stream>>>(fc_w, INsz, 0, ws_fcw, H0sz, INsz, INsz, H0P * INsz);
    k_cvt_pad<<<nblk(RNP * H0sz, TPB), TPB, 0, stream>>>(x2in_w, H0sz, 0, ws_x2in, Rsz, H0sz, H0sz, RNP * H0sz);
    k_cvt_pad<<<nblk(RNP * RP, TPB), TPB, 0, stream>>>(rec4in_w,  Rsz, 0, ws_rec4in,  Rsz, Rsz, RP, RNP * RP);
    k_cvt_pad<<<nblk(RNP * RP, TPB), TPB, 0, stream>>>(in2out_w,  Rsz, 0, ws_in2out,  Rsz, Rsz, RP, RNP * RP);
    k_cvt_pad<<<nblk(RNP * RP, TPB), TPB, 0, stream>>>(rec4out_w, Rsz, 0, ws_rec4out, Rsz, Rsz, RP, RNP * RP);
    k_cvt_pad<<<nblk(RNP * RP, TPB), TPB, 0, stream>>>(out2in_w,  Rsz, 0, ws_out2in,  Rsz, Rsz, RP, RNP * RP);
    k_cvt_pad<<<nblk(Bsz * RP, TPB), TPB, 0, stream>>>(spk2, H1sz, 0,   ws_spkout, Bsz, Rsz, RP, Bsz * RP);
    k_cvt_pad<<<nblk(Bsz * RP, TPB), TPB, 0, stream>>>(spk2, H1sz, Rsz, ws_spkin,  Bsz, Rsz, RP, Bsz * RP);

    // --- Layer 1: (4096/128) x (4096/64) = 32 x 64 blocks ---
    k_gemm1<<<32 * 64, TPB, 0, stream>>>(ws_xt, ws_fcw, fc_b, mem1, spk1, b1,
                                         tau_m1, tau_adp1,
                                         o_mem1n, o_spk1n, o_b1n, ws_spk1n);

    // --- r_out_: 32 x (2048/64)=32 blocks ---
    k_rout<<<32 * 32, TPB, 0, stream>>>(ws_spkout, ws_spkin, ws_rec4out, ws_in2out,
                                        rec4out_b, in2out_b,
                                        mem2, spk2, b2, tau_m2, tau_adp2,
                                        o_mem2n, o_spk2n, o_b2n);

    // --- r_in_: 32 x 32 blocks ---
    k_rin<<<32 * 32, TPB, 0, stream>>>(ws_spk1n, ws_x2in, ws_spkin, ws_rec4in,
                                       ws_spkout, ws_out2in,
                                       x2in_b, rec4in_b, out2in_b,
                                       mem2, spk2, b2, tau_m2, tau_adp2,
                                       o_mem2n, o_spk2n, o_b2n);

    // --- Output head ---
    k_outsum<<<nblk(Bsz * OUTsz, TPB), TPB, 0, stream>>>(o_spk2n, mem_out, out_tau_m, o_memoutn);
    k_logsm<<<nblk(Bsz, TPB), TPB, 0, stream>>>(o_memoutn, o_logsm);
}
